// MeshConvolution_49538152792831
// MI455X (gfx1250) — matrix operand order repaired and checked by reading, not compiled, not run on hardware
//
#include <hip/hip_runtime.h>

#define NPTS 65536
#define NB 8

typedef __attribute__((ext_vector_type(16))) __bf16 v16bf;
typedef __attribute__((ext_vector_type(8)))  __bf16 v8bf;
typedef __attribute__((ext_vector_type(8)))  float  v8f;

union AB16 { v16bf v; v8bf h[2]; };

// A fragment (16x32 bf16, M x K). Lanes 0-15 rows, halves map to
// K = {0..7,16..23} (+8 for lanes 16-31), as two contiguous 8-half chunks.
__device__ inline v16bf lds_a(const __bf16* W, int ldw, int m0, int kt, int lane) {
    int m = m0 + (lane & 15);
    int koff = kt * 32 + ((lane & 16) ? 8 : 0);
    const __bf16* p = W + m * ldw + koff;
    AB16 a;
    a.h[0] = *(const v8bf*)(p);
    a.h[1] = *(const v8bf*)(p + 16);
    return a.v;
}

// B fragment (32x16 bf16, K x N). Lane = column, halves map to
// K = {0..7,16..23} (+8 for lanes 16-31), as two contiguous 8-half chunks.
__device__ inline v16bf lds_b(const __bf16* X, int ldx, int col, int kt, int lane) {
    int koff = kt * 32 + ((lane & 16) ? 8 : 0);
    const __bf16* p = X + col * ldx + koff;
    AB16 b;
    b.h[0] = *(const v8bf*)(p);
    b.h[1] = *(const v8bf*)(p + 16);
    return b.v;
}

__device__ inline v8f wmma_bf16(v16bf a, v16bf b, v8f c) {
    return __builtin_amdgcn_wmma_f32_16x16x32_bf16(false, a, false, b, (short)0, c, false, false);
}

__device__ inline float redu16(float x) {
    #pragma unroll
    for (int off = 1; off < 16; off <<= 1) x += __shfl_xor(x, off, 16);
    return x;
}

__global__ __launch_bounds__(256) void k_zero_stats(float* ws) {
    int i = blockIdx.x * blockDim.x + threadIdx.x;
    if (i < 2048) ws[i] = 0.0f;
}

// ---------------- BN finalize: per-channel scale/shift ---------------------
__global__ __launch_bounds__(256) void k_finalize(
    const float* __restrict__ sum, const float* __restrict__ sq,
    const float* __restrict__ gamma, const float* __restrict__ beta,
    float* __restrict__ scale, float* __restrict__ shift, int C, float inv_cnt)
{
    int c = blockIdx.x * blockDim.x + threadIdx.x;
    if (c < C) {
        float mu  = sum[c] * inv_cnt;
        float var = sq[c] * inv_cnt - mu * mu;
        float sc  = gamma[c] * rsqrtf(var + 1e-5f);
        scale[c] = sc;
        shift[c] = beta[c] - mu * sc;
    }
}

// ---------------- Stage 1: W_comb[128x128] @ concat(spatial,structural) ----
// pass 0: GEMM -> per-channel sum/sumsq only (nothing stored to HBM).
// pass 1: recompute GEMM, fuse BN+ReLU, nontemporal-store final sp once.
__global__ __launch_bounds__(256) void k_gemm_comb(
    const float* __restrict__ spat, const float* __restrict__ stru,
    const float* __restrict__ Wc, const float* __restrict__ bias,
    float* __restrict__ sum, float* __restrict__ sq,
    const float* __restrict__ scale, const float* __restrict__ shift,
    float* __restrict__ outFinal, int pass)
{
    __shared__ __bf16 Wl[128 * 136];
    __shared__ __bf16 Xl[64 * 136];
    const int tid = threadIdx.x, lane = tid & 31, wave = tid >> 5;
    const int blocksPerB = NPTS / 64;
    const int b  = blockIdx.x / blocksPerB;
    const int n0 = (blockIdx.x % blocksPerB) * 64;

    for (int i = tid; i < 128 * 128; i += 256) {
        int r = i >> 7, c = i & 127;
        Wl[r * 136 + c] = (__bf16)Wc[i];
    }
    for (int i = tid; i < 128 * 64; i += 256) {
        int c = i >> 6, q = i & 63;
        const float* s = (c < 64) ? spat : stru;
        float v = s[((long)b * 64 + (c & 63)) * NPTS + n0 + q];
        Xl[q * 136 + c] = (__bf16)v;
    }
    __syncthreads();

    const int m0 = wave * 16;
    const v8f vz = {0.f,0.f,0.f,0.f,0.f,0.f,0.f,0.f};
    v8f acc[4];
    #pragma unroll
    for (int t = 0; t < 4; ++t) acc[t] = vz;

    #pragma unroll
    for (int kt = 0; kt < 4; ++kt) {
        v16bf a = lds_a(Wl, 136, m0, kt, lane);
        #pragma unroll
        for (int ct = 0; ct < 4; ++ct) {
            v16bf bb = lds_b(Xl, 136, ct * 16 + (lane & 15), kt, lane);
            acc[ct] = wmma_bf16(a, bb, acc[ct]);
        }
    }

    const int hi = (lane >> 4) & 1;
    if (pass == 0) {
        float rs[8], rq[8];
        #pragma unroll
        for (int v = 0; v < 8; ++v) { rs[v] = 0.f; rq[v] = 0.f; }
        #pragma unroll
        for (int ct = 0; ct < 4; ++ct) {
            #pragma unroll
            for (int v = 0; v < 8; ++v) {
                float val = acc[ct][v] + bias[m0 + v + hi * 8];
                rs[v] += val; rq[v] += val * val;
            }
        }
        #pragma unroll
        for (int v = 0; v < 8; ++v) {
            float a0 = redu16(rs[v]);
            float a1 = redu16(rq[v]);
            if ((lane & 15) == 0) {
                int m = m0 + v + hi * 8;
                atomicAdd(&sum[m], a0);
                atomicAdd(&sq[m], a1);
            }
        }
    } else {
        #pragma unroll
        for (int ct = 0; ct < 4; ++ct) {
            #pragma unroll
            for (int v = 0; v < 8; ++v) {
                int m = m0 + v + hi * 8;
                float val = acc[ct][v] + bias[m];
                val = fmaxf(val * scale[m] + shift[m], 0.f);
                int n = n0 + ct * 16 + (lane & 15);
                // Streamed output, never re-read: NT keeps L2 for the gather set.
                __builtin_nontemporal_store(val, &outFinal[((long)b * 128 + m) * NPTS + n]);
            }
        }
    }
}

// ---------------- Stage 2: gather + W_cat[64x128], two passes --------------
// pass 0: accumulate BN stats over (b,n,k). pass 1: recompute, BN+ReLU,
// max over k=3, store bf16 st_mid. Gathers hit the 192MB L2 (feat = 128MiB).
__global__ __launch_bounds__(256) void k_gemm_cat(
    const float* __restrict__ stru, const int* __restrict__ nidx,
    const float* __restrict__ Wc, const float* __restrict__ bias,
    float* __restrict__ sum, float* __restrict__ sq,
    const float* __restrict__ scale, const float* __restrict__ shift,
    __bf16* __restrict__ stMid, int pass)
{
    __shared__ __bf16 Wl[64 * 136];
    __shared__ __bf16 Xl[96 * 136];
    __shared__ __bf16 Yl[64 * 96];
    __shared__ int gcol[96];

    const int tid = threadIdx.x, lane = tid & 31, wave = tid >> 5;
    const int blocksPerB = NPTS / 32;
    const int b  = blockIdx.x / blocksPerB;
    const int t0 = (blockIdx.x % blocksPerB) * 32;   // triple (face) base

    for (int i = tid; i < 64 * 128; i += 256) {
        int r = i >> 7, c = i & 127;
        Wl[r * 136 + c] = (__bf16)Wc[i];
    }
    if (tid < 96) {
        int n = t0 + tid / 3, k = tid % 3;
        gcol[tid] = nidx[((long)b * NPTS + n) * 3 + k];
    }
    __syncthreads();
    for (int i = tid; i < 96 * 128; i += 256) {
        int q = i % 96, c = i / 96;
        int n = t0 + q / 3;
        float v = (c < 64) ? stru[((long)b * 64 + c) * NPTS + n]
                           : stru[((long)b * 64 + (c - 64)) * NPTS + gcol[q]];
        Xl[q * 136 + c] = (__bf16)v;
    }
    __syncthreads();

    const int rowt = wave & 3, chalf = wave >> 2;
    const int m0 = rowt * 16;
    const v8f vz = {0.f,0.f,0.f,0.f,0.f,0.f,0.f,0.f};
    v8f acc[3];
    #pragma unroll
    for (int t = 0; t < 3; ++t) acc[t] = vz;

    #pragma unroll
    for (int kt = 0; kt < 4; ++kt) {
        v16bf a = lds_a(Wl, 136, m0, kt, lane);
        #pragma unroll
        for (int ct = 0; ct < 3; ++ct) {
            int col = chalf * 48 + ct * 16 + (lane & 15);
            v16bf bb = lds_b(Xl, 136, col, kt, lane);
            acc[ct] = wmma_bf16(a, bb, acc[ct]);
        }
    }

    const int hi = (lane >> 4) & 1;
    if (pass == 0) {
        float rs[8], rq[8];
        #pragma unroll
        for (int v = 0; v < 8; ++v) { rs[v] = 0.f; rq[v] = 0.f; }
        #pragma unroll
        for (int ct = 0; ct < 3; ++ct) {
            #pragma unroll
            for (int v = 0; v < 8; ++v) {
                float val = acc[ct][v] + bias[m0 + v + hi * 8];
                rs[v] += val; rq[v] += val * val;
            }
        }
        #pragma unroll
        for (int v = 0; v < 8; ++v) {
            float a0 = redu16(rs[v]);
            float a1 = redu16(rq[v]);
            if ((lane & 15) == 0) {
                int m = m0 + v + hi * 8;
                atomicAdd(&sum[m], a0);
                atomicAdd(&sq[m], a1);
            }
        }
    } else {
        #pragma unroll
        for (int ct = 0; ct < 3; ++ct) {
            #pragma unroll
            for (int v = 0; v < 8; ++v) {
                int m = m0 + v + hi * 8;
                float val = acc[ct][v] + bias[m];
                val = fmaxf(val * scale[m] + shift[m], 0.f);
                int col = chalf * 48 + ct * 16 + (lane & 15);
                Yl[m * 96 + col] = (__bf16)val;
            }
        }
        __syncthreads();
        for (int o = tid; o < 64 * 32; o += 256) {
            int r = o >> 5, t = o & 31;
            float x0 = (float)Yl[r * 96 + 3 * t];
            float x1 = (float)Yl[r * 96 + 3 * t + 1];
            float x2 = (float)Yl[r * 96 + 3 * t + 2];
            // Re-read by stage 3 twice: keep cacheable (64 MiB fits L2).
            stMid[((long)b * 64 + r) * NPTS + (t0 + t)] = (__bf16)fmaxf(x0, fmaxf(x1, x2));
        }
    }
}

// ---------------- Stage 3: W_agg[128x64] @ st_mid (bf16 in ws) -------------
// pass 0: stats. pass 1: recompute (st_mid now L2-resident), fuse BN+ReLU,
// nontemporal-store final st once.
__global__ __launch_bounds__(256) void k_gemm_agg(
    const __bf16* __restrict__ stMid, const float* __restrict__ Wa,
    const float* __restrict__ bias,
    float* __restrict__ sum, float* __restrict__ sq,
    const float* __restrict__ scale, const float* __restrict__ shift,
    float* __restrict__ outFinal, int pass)
{
    __shared__ __bf16 Wl[128 * 72];
    __shared__ __bf16 Xl[128 * 72];
    const int tid = threadIdx.x, lane = tid & 31, wave = tid >> 5;
    const int blocksPerB = NPTS / 128;
    const int b  = blockIdx.x / blocksPerB;
    const int n0 = (blockIdx.x % blocksPerB) * 128;

    for (int i = tid; i < 128 * 64; i += 256) {
        int r = i >> 6, c = i & 63;
        Wl[r * 72 + c] = (__bf16)Wa[i];
    }
    for (int i = tid; i < 128 * 64; i += 256) {
        int q = i & 127, c = i >> 7;
        Xl[q * 72 + c] = stMid[((long)b * 64 + c) * NPTS + n0 + q];
    }
    __syncthreads();

    const int m0 = wave * 16;
    const v8f vz = {0.f,0.f,0.f,0.f,0.f,0.f,0.f,0.f};
    v8f acc[8];
    #pragma unroll
    for (int t = 0; t < 8; ++t) acc[t] = vz;

    #pragma unroll
    for (int kt = 0; kt < 2; ++kt) {
        v16bf a = lds_a(Wl, 72, m0, kt, lane);
        #pragma unroll
        for (int ct = 0; ct < 8; ++ct) {
            v16bf bb = lds_b(Xl, 72, ct * 16 + (lane & 15), kt, lane);
            acc[ct] = wmma_bf16(a, bb, acc[ct]);
        }
    }

    const int hi = (lane >> 4) & 1;
    if (pass == 0) {
        float rs[8], rq[8];
        #pragma unroll
        for (int v = 0; v < 8; ++v) { rs[v] = 0.f; rq[v] = 0.f; }
        #pragma unroll
        for (int ct = 0; ct < 8; ++ct) {
            #pragma unroll
            for (int v = 0; v < 8; ++v) {
                float val = acc[ct][v] + bias[m0 + v + hi * 8];
                rs[v] += val; rq[v] += val * val;
            }
        }
        #pragma unroll
        for (int v = 0; v < 8; ++v) {
            float a0 = redu16(rs[v]);
            float a1 = redu16(rq[v]);
            if ((lane & 15) == 0) {
                int m = m0 + v + hi * 8;
                atomicAdd(&sum[m], a0);
                atomicAdd(&sq[m], a1);
            }
        }
    } else {
        #pragma unroll
        for (int ct = 0; ct < 8; ++ct) {
            #pragma unroll
            for (int v = 0; v < 8; ++v) {
                int m = m0 + v + hi * 8;
                float val = acc[ct][v] + bias[m];
                val = fmaxf(val * scale[m] + shift[m], 0.f);
                int n = n0 + ct * 16 + (lane & 15);
                __builtin_nontemporal_store(val, &outFinal[((long)b * 128 + m) * NPTS + n]);
            }
        }
    }
}

extern "C" void kernel_launch(void* const* d_in, const int* in_sizes, int n_in,
                              void* d_out, int out_size, void* d_ws, size_t ws_size,
                              hipStream_t stream) {
    const float* spat   = (const float*)d_in[0];
    const float* stru   = (const float*)d_in[1];
    const int*   nidx   = (const int*)d_in[2];
    const float* Wcomb  = (const float*)d_in[3];
    const float* bcomb  = (const float*)d_in[4];
    const float* gcomb  = (const float*)d_in[5];
    const float* becomb = (const float*)d_in[6];
    const float* Wcat   = (const float*)d_in[7];
    const float* bcat   = (const float*)d_in[8];
    const float* gcat   = (const float*)d_in[9];
    const float* becat  = (const float*)d_in[10];
    const float* Wagg   = (const float*)d_in[11];
    const float* bagg   = (const float*)d_in[12];
    const float* gagg   = (const float*)d_in[13];
    const float* beagg  = (const float*)d_in[14];

    float* out = (float*)d_out;
    float* ws  = (float*)d_ws;
    float* sum1 = ws;        float* sq1 = ws + 128;
    float* sum2 = ws + 256;  float* sq2 = ws + 320;
    float* sum3 = ws + 384;  float* sq3 = ws + 512;
    float* scale1 = ws + 640;  float* shift1 = ws + 768;
    float* scale2 = ws + 896;  float* shift2 = ws + 960;
    float* scale3 = ws + 1024; float* shift3 = ws + 1152;
    __bf16* stMid = (__bf16*)((char*)d_ws + 8192);   // B*64*N bf16 = 64 MiB

    float* sp = out;
    float* st = out + (long)NB * 128 * NPTS;

    k_zero_stats<<<8, 256, 0, stream>>>(ws);

    // Stage 1: stats pass -> finalize -> fused BN+ReLU apply pass (NT store)
    k_gemm_comb<<<NB * (NPTS / 64), 256, 0, stream>>>(
        spat, stru, Wcomb, bcomb, sum1, sq1, scale1, shift1, sp, 0);
    k_finalize<<<1, 128, 0, stream>>>(sum1, sq1, gcomb, becomb, scale1, shift1, 128,
                                      1.0f / ((float)NB * (float)NPTS));
    k_gemm_comb<<<NB * (NPTS / 64), 256, 0, stream>>>(
        spat, stru, Wcomb, bcomb, sum1, sq1, scale1, shift1, sp, 1);

    // Stage 2: stats pass -> finalize -> apply pass (BN+ReLU+max_k -> bf16 ws)
    k_gemm_cat<<<NB * (NPTS / 32), 256, 0, stream>>>(
        stru, nidx, Wcat, bcat, sum2, sq2, scale2, shift2, stMid, 0);
    k_finalize<<<1, 64, 0, stream>>>(sum2, sq2, gcat, becat, scale2, shift2, 64,
                                     1.0f / (3.0f * (float)NB * (float)NPTS));
    k_gemm_cat<<<NB * (NPTS / 32), 256, 0, stream>>>(
        stru, nidx, Wcat, bcat, sum2, sq2, scale2, shift2, stMid, 1);

    // Stage 3: stats pass -> finalize -> fused BN+ReLU apply pass (NT store)
    k_gemm_agg<<<NB * (NPTS / 128), 256, 0, stream>>>(
        stMid, Wagg, bagg, sum3, sq3, scale3, shift3, st, 0);
    k_finalize<<<1, 128, 0, stream>>>(sum3, sq3, gagg, beagg, scale3, shift3, 128,
                                      1.0f / ((float)NB * (float)NPTS));
    k_gemm_agg<<<NB * (NPTS / 128), 256, 0, stream>>>(
        stMid, Wagg, bagg, sum3, sq3, scale3, shift3, st, 1);
}
